// StackedLSTM_13537736917491
// MI455X (gfx1250) — matrix operand order repaired and checked by reading, not compiled, not run on hardware
//
#include <hip/hip_runtime.h>

typedef __attribute__((ext_vector_type(16))) __bf16 v16bf;
typedef __attribute__((ext_vector_type(8)))  __bf16 v8bf;
typedef __attribute__((ext_vector_type(8)))  float  v8f;

#define SEQ   1024
#define BATCH 128
#define NIN   256
#define HID   512
#define GATES 2048      // 4*HID
#define K0    768       // NIN + HID   (layer 0)
#define K1    1024      // HID + HID   (layer 1)
#define KT0   24        // K0/32
#define KT1   32        // K1/32
#define NT    128       // GATES/16
#define LDK0  (K0 + 8)  // padded LDS row stride (bank-conflict-free)
#define LDK1  (K1 + 8)

__device__ __forceinline__ float sigmoidf_(float x) {
  return 1.0f / (1.0f + __expf(-x));
}

// ---------------------------------------------------------------------------
// Pack weights into WMMA-B fragment order.
// B-matrix tile (kt, nt) covers K=[kt*32, kt*32+32), N=[nt*16, nt*16+16).
// Order used here: lanes 0-15 hold column N=lane with K 0..15 of the tile,
//          lanes 16-31 hold column N=lane-16 with K 16..31.
// Stored as 32 lanes * 16 bf16 contiguous per lane -> one 32B load per lane.
// ---------------------------------------------------------------------------
__global__ __launch_bounds__(256) void pack_w0(const float* __restrict__ Wih,
                                               const float* __restrict__ Whh,
                                               __bf16* __restrict__ Wp) {
  int tid  = blockIdx.x * 256 + threadIdx.x;      // KT0*NT*32 threads
  int tile = tid >> 5, lane = tid & 31;
  int kt = tile / NT, nt = tile - kt * NT;
  int n     = nt * 16 + (lane & 15);
  int kbase = kt * 32 + (lane >> 4) * 16;
  v16bf vals;
#pragma unroll
  for (int i = 0; i < 16; ++i) {
    int k = kbase + i;
    float v = (k < NIN) ? Wih[(size_t)n * NIN + k]
                        : Whh[(size_t)n * HID + (k - NIN)];
    vals[i] = (__bf16)v;
  }
  *(v16bf*)(Wp + (size_t)tile * 512 + lane * 16) = vals;
}

__global__ __launch_bounds__(256) void pack_w1(const float* __restrict__ Wih,
                                               const float* __restrict__ Whh,
                                               __bf16* __restrict__ Wp) {
  int tid  = blockIdx.x * 256 + threadIdx.x;      // KT1*NT*32 threads
  int tile = tid >> 5, lane = tid & 31;
  int kt = tile / NT, nt = tile - kt * NT;
  int n     = nt * 16 + (lane & 15);
  int kbase = kt * 32 + (lane >> 4) * 16;
  v16bf vals;
#pragma unroll
  for (int i = 0; i < 16; ++i) {
    int k = kbase + i;
    float v = (k < HID) ? Wih[(size_t)n * HID + k]
                        : Whh[(size_t)n * HID + (k - HID)];
    vals[i] = (__bf16)v;
  }
  *(v16bf*)(Wp + (size_t)tile * 512 + lane * 16) = vals;
}

// Fuse biases, zero initial h/c state (re-done every call -> deterministic).
__global__ __launch_bounds__(256) void init_state(
    const float* __restrict__ bih0, const float* __restrict__ bhh0,
    const float* __restrict__ bih1, const float* __restrict__ bhh1,
    float* __restrict__ b0, float* __restrict__ b1,
    __bf16* __restrict__ h0, __bf16* __restrict__ h1,
    float* __restrict__ c0, float* __restrict__ c1) {
  int i = blockIdx.x * 256 + threadIdx.x;         // BATCH*HID threads
  if (i < GATES) {
    b0[i] = bih0[i] + bhh0[i];
    b1[i] = bih1[i] + bhh1[i];
  }
  c0[i] = 0.0f;
  c1[i] = 0.0f;
  h0[i] = (__bf16)0.0f;
  h1[i] = (__bf16)0.0f;
}

// ---------------------------------------------------------------------------
// Layer-0 step: gates = [x_s | h_prev] * Wp0 + b0 ; fused LSTM cell.
// 32 blocks x 8 waves; block -> (mtile = blk>>2), wave -> ncol slice of HID.
// Each wave computes the i/f/g/o 16x16 tiles for its (mtile, ncol) so the
// cell update is wave-local.
// ---------------------------------------------------------------------------
__global__ __launch_bounds__(256) void lstm_step_l0(
    const float* __restrict__ x,        // [BATCH, NIN]   (this step)
    const __bf16* __restrict__ hprev,   // [BATCH, HID]
    const __bf16* __restrict__ Wp,      // packed tiles
    const float* __restrict__ bias,     // [GATES] fused
    float* __restrict__ c,              // [BATCH, HID] in-place
    __bf16* __restrict__ hnext)         // [BATCH, HID]
{
  __shared__ __attribute__((aligned(16))) __bf16 zl[16 * LDK0];
  const int t    = threadIdx.x;
  const int wave = t >> 5, lane = t & 31;
  const int mtile   = blockIdx.x >> 2;
  const int ncol    = (blockIdx.x & 3) * 8 + wave;
  const int rowbase = mtile * 16;

  // Stage A rows into LDS: x (fp32 -> bf16) then h_prev (bf16 copy, 16B vec).
  for (int idx = t; idx < 16 * NIN; idx += 256) {
    int m = idx >> 8, k = idx & (NIN - 1);
    zl[m * LDK0 + k] = (__bf16)x[(size_t)(rowbase + m) * NIN + k];
  }
  {
    const uint4* src = (const uint4*)(hprev + (size_t)rowbase * HID);
    for (int idx = t; idx < 16 * 64; idx += 256) {
      int m = idx >> 6, q = idx & 63;
      *(uint4*)(zl + m * LDK0 + NIN + q * 8) = src[m * 64 + q];
    }
  }
  __syncthreads();

  v8f ai = {}, af = {}, ag = {}, ao = {};
  const int m  = lane & 15;
  const int hi = lane >> 4;
  const __bf16* arow = zl + m * LDK0 + hi * 16;
  const __bf16* wl   = Wp + (size_t)lane * 16;

  for (int kt = 0; kt < KT0; ++kt) {
    v8bf alo = *(const v8bf*)(arow + kt * 32);
    v8bf ahi = *(const v8bf*)(arow + kt * 32 + 8);
    v16bf a;
#pragma unroll
    for (int i = 0; i < 8; ++i) { a[i] = alo[i]; a[i + 8] = ahi[i]; }
    const __bf16* wt = wl + (size_t)(kt * NT) * 512;
    v16bf bI = *(const v16bf*)(wt + (size_t)(ncol)      * 512);
    v16bf bF = *(const v16bf*)(wt + (size_t)(32 + ncol) * 512);
    v16bf bG = *(const v16bf*)(wt + (size_t)(64 + ncol) * 512);
    v16bf bO = *(const v16bf*)(wt + (size_t)(96 + ncol) * 512);
    ai = __builtin_amdgcn_wmma_f32_16x16x32_bf16(false, a, false, bI, (short)0, ai, false, false);
    af = __builtin_amdgcn_wmma_f32_16x16x32_bf16(false, a, false, bF, (short)0, af, false, false);
    ag = __builtin_amdgcn_wmma_f32_16x16x32_bf16(false, a, false, bG, (short)0, ag, false, false);
    ao = __builtin_amdgcn_wmma_f32_16x16x32_bf16(false, a, false, bO, (short)0, ao, false, false);
  }

  // Epilogue: LSTM cell. D layout: VGPR r -> M = r + 8*hi, lane&15 -> N.
  const int col = ncol * 16 + m;
  const float Bi = bias[col], Bf = bias[HID + col];
  const float Bg = bias[2 * HID + col], Bo = bias[3 * HID + col];
  const int r0 = rowbase + hi * 8;
#pragma unroll
  for (int r = 0; r < 8; ++r) {
    size_t o = (size_t)(r0 + r) * HID + col;
    float gi = ai[r] + Bi, gf = af[r] + Bf, gg = ag[r] + Bg, go = ao[r] + Bo;
    float cn = sigmoidf_(gf) * c[o] + sigmoidf_(gi) * tanhf(gg);
    c[o] = cn;
    hnext[o] = (__bf16)(sigmoidf_(go) * tanhf(cn));
  }
}

// ---------------------------------------------------------------------------
// Layer-1 step: gates = [h0_cur | h1_prev] * Wp1 + b1 ; writes h1 (bf16) and
// the fp32 sequence output for this step.
// ---------------------------------------------------------------------------
__global__ __launch_bounds__(256) void lstm_step_l1(
    const __bf16* __restrict__ h0cur,   // [BATCH, HID]
    const __bf16* __restrict__ h1prev,  // [BATCH, HID]
    const __bf16* __restrict__ Wp,
    const float* __restrict__ bias,
    float* __restrict__ c,
    __bf16* __restrict__ hnext,
    float* __restrict__ out)            // d_out + s*BATCH*HID
{
  __shared__ __attribute__((aligned(16))) __bf16 zl[16 * LDK1];
  const int t    = threadIdx.x;
  const int wave = t >> 5, lane = t & 31;
  const int mtile   = blockIdx.x >> 2;
  const int ncol    = (blockIdx.x & 3) * 8 + wave;
  const int rowbase = mtile * 16;

  {
    const uint4* s0 = (const uint4*)(h0cur  + (size_t)rowbase * HID);
    const uint4* s1 = (const uint4*)(h1prev + (size_t)rowbase * HID);
    for (int idx = t; idx < 16 * 64; idx += 256) {
      int m = idx >> 6, q = idx & 63;
      *(uint4*)(zl + m * LDK1 + q * 8)       = s0[m * 64 + q];
      *(uint4*)(zl + m * LDK1 + HID + q * 8) = s1[m * 64 + q];
    }
  }
  __syncthreads();

  v8f ai = {}, af = {}, ag = {}, ao = {};
  const int m  = lane & 15;
  const int hi = lane >> 4;
  const __bf16* arow = zl + m * LDK1 + hi * 16;
  const __bf16* wl   = Wp + (size_t)lane * 16;

  for (int kt = 0; kt < KT1; ++kt) {
    v8bf alo = *(const v8bf*)(arow + kt * 32);
    v8bf ahi = *(const v8bf*)(arow + kt * 32 + 8);
    v16bf a;
#pragma unroll
    for (int i = 0; i < 8; ++i) { a[i] = alo[i]; a[i + 8] = ahi[i]; }
    const __bf16* wt = wl + (size_t)(kt * NT) * 512;
    v16bf bI = *(const v16bf*)(wt + (size_t)(ncol)      * 512);
    v16bf bF = *(const v16bf*)(wt + (size_t)(32 + ncol) * 512);
    v16bf bG = *(const v16bf*)(wt + (size_t)(64 + ncol) * 512);
    v16bf bO = *(const v16bf*)(wt + (size_t)(96 + ncol) * 512);
    ai = __builtin_amdgcn_wmma_f32_16x16x32_bf16(false, a, false, bI, (short)0, ai, false, false);
    af = __builtin_amdgcn_wmma_f32_16x16x32_bf16(false, a, false, bF, (short)0, af, false, false);
    ag = __builtin_amdgcn_wmma_f32_16x16x32_bf16(false, a, false, bG, (short)0, ag, false, false);
    ao = __builtin_amdgcn_wmma_f32_16x16x32_bf16(false, a, false, bO, (short)0, ao, false, false);
  }

  const int col = ncol * 16 + m;
  const float Bi = bias[col], Bf = bias[HID + col];
  const float Bg = bias[2 * HID + col], Bo = bias[3 * HID + col];
  const int r0 = rowbase + hi * 8;
#pragma unroll
  for (int r = 0; r < 8; ++r) {
    size_t o = (size_t)(r0 + r) * HID + col;
    float gi = ai[r] + Bi, gf = af[r] + Bf, gg = ag[r] + Bg, go = ao[r] + Bo;
    float cn = sigmoidf_(gf) * c[o] + sigmoidf_(gi) * tanhf(gg);
    c[o] = cn;
    float hv = sigmoidf_(go) * tanhf(cn);
    hnext[o] = (__bf16)hv;
    out[o]   = hv;
  }
}

// ---------------------------------------------------------------------------
extern "C" void kernel_launch(void* const* d_in, const int* in_sizes, int n_in,
                              void* d_out, int out_size, void* d_ws, size_t ws_size,
                              hipStream_t stream) {
  const float* inputs = (const float*)d_in[0];   // [SEQ, BATCH, NIN]
  const float* W_ih0  = (const float*)d_in[1];
  const float* W_hh0  = (const float*)d_in[2];
  const float* b_ih0  = (const float*)d_in[3];
  const float* b_hh0  = (const float*)d_in[4];
  const float* W_ih1  = (const float*)d_in[5];
  const float* W_hh1  = (const float*)d_in[6];
  const float* b_ih1  = (const float*)d_in[7];
  const float* b_hh1  = (const float*)d_in[8];
  float* out = (float*)d_out;

  // Workspace carve (all offsets 16B-aligned; total ~8.0 MB).
  char* p = (char*)d_ws;
  __bf16* Wp0 = (__bf16*)p;  p += (size_t)KT0 * NT * 512 * 2;  // 3 MB
  __bf16* Wp1 = (__bf16*)p;  p += (size_t)KT1 * NT * 512 * 2;  // 4 MB
  float*  b0  = (float*)p;   p += GATES * 4;
  float*  b1  = (float*)p;   p += GATES * 4;
  __bf16* h0buf = (__bf16*)p; p += 2 * (size_t)BATCH * HID * 2;
  __bf16* h1buf = (__bf16*)p; p += 2 * (size_t)BATCH * HID * 2;
  float*  c0  = (float*)p;   p += (size_t)BATCH * HID * 4;
  float*  c1  = (float*)p;   p += (size_t)BATCH * HID * 4;

  // One-time (per call) prep: pack weights, fuse biases, zero state.
  pack_w0<<<(KT0 * NT * 32) / 256, 256, 0, stream>>>(W_ih0, W_hh0, Wp0);
  pack_w1<<<(KT1 * NT * 32) / 256, 256, 0, stream>>>(W_ih1, W_hh1, Wp1);
  init_state<<<(BATCH * HID) / 256, 256, 0, stream>>>(
      b_ih0, b_hh0, b_ih1, b_hh1, b0, b1,
      h0buf /*buf 0*/, h1buf /*buf 0*/, c0, c1);

  const size_t hstate = (size_t)BATCH * HID;
  for (int s = 0; s < SEQ; ++s) {
    const float* xs = inputs + (size_t)s * BATCH * NIN;
    __bf16* h0p = h0buf + (size_t)(s & 1) * hstate;
    __bf16* h0n = h0buf + (size_t)((s + 1) & 1) * hstate;
    __bf16* h1p = h1buf + (size_t)(s & 1) * hstate;
    __bf16* h1n = h1buf + (size_t)((s + 1) & 1) * hstate;
    lstm_step_l0<<<32, 256, 0, stream>>>(xs, h0p, Wp0, b0, c0, h0n);
    lstm_step_l1<<<32, 256, 0, stream>>>(h0n, h1p, Wp1, b1, c1, h1n,
                                         out + (size_t)s * hstate);
  }
}
